// PRGNN_26336739459481
// MI455X (gfx1250) — matrix operand order repaired and checked by reading, not compiled, not run on hardware
//
#include <hip/hip_runtime.h>
#include <hip/hip_bf16.h>
#include <stdint.h>

#define NNODES 50000
#define NEDGES 800000

typedef __bf16 v16bf __attribute__((ext_vector_type(16)));
typedef float  v8f   __attribute__((ext_vector_type(8)));

__device__ __forceinline__ unsigned short f2bf(float x) {
    unsigned u = __builtin_bit_cast(unsigned, x);
    u += 0x7FFFu + ((u >> 16) & 1u);            // round-to-nearest-even
    return (unsigned short)(u >> 16);
}
__device__ __forceinline__ float bf2f(unsigned short h) {
    unsigned u = ((unsigned)h) << 16;
    return __builtin_bit_cast(float, u);
}

// CDNA5 async global->LDS copy: 16B per lane, tracked by ASYNCcnt.
__device__ __forceinline__ void async_copy_b128(unsigned lds_addr, const void* gaddr) {
    asm volatile("global_load_async_to_lds_b128 %0, %1, off"
                 :: "v"(lds_addr), "v"((unsigned long long)(uintptr_t)gaddr)
                 : "memory");
}
__device__ __forceinline__ void wait_async0() {
    asm volatile("s_wait_asynccnt 0x0" ::: "memory");
}

// ---------------- elementwise / prep kernels ----------------

__global__ void k_zero(float* p, long n) {
    long stride = (long)gridDim.x * blockDim.x;
    for (long i = (long)blockIdx.x * blockDim.x + threadIdx.x; i < n; i += stride)
        p[i] = 0.0f;
}

__global__ void k_cvt_bf16(const float* s, unsigned short* d, long n) {
    long stride = (long)gridDim.x * blockDim.x;
    for (long i = (long)blockIdx.x * blockDim.x + threadIdx.x; i < n; i += stride)
        d[i] = f2bf(s[i]);
}

// convert weight [R,C] f32 -> [Rp,Cp] bf16 with zero padding
__global__ void k_padcvt(const float* s, unsigned short* d, int R, int C, int Rp, int Cp) {
    int total = Rp * Cp;
    for (int i = blockIdx.x * blockDim.x + threadIdx.x; i < total; i += gridDim.x * blockDim.x) {
        int r = i / Cp, c = i % Cp;
        float v = (r < R && c < C) ? s[r * C + c] : 0.0f;
        d[i] = f2bf(v);
    }
}

__global__ void k_padbias(const float* s, float* d, int n, int np) {
    int i = threadIdx.x;
    if (i < np) d[i] = (i < n) ? s[i] : 0.0f;
}

// ---------------- graph aggregation kernels ----------------

// degree count (float; exact for small counts)
__global__ void k_count(const int* ei, float* cnt) {
    for (int e = blockIdx.x * blockDim.x + threadIdx.x; e < NEDGES; e += gridDim.x * blockDim.x)
        atomicAdd(&cnt[ei[NEDGES + e]], 1.0f);
}

// agg[dst, f] += feat[src, f]   (feat bf16, agg f32). F = 1<<Fsh.
__global__ void k_scatter(const unsigned short* feat, const int* ei, float* agg, int Fsh) {
    const int F = 1 << Fsh;
    const long total = (long)NEDGES << Fsh;
    long stride = (long)gridDim.x * blockDim.x;
    for (long i = (long)blockIdx.x * blockDim.x + threadIdx.x; i < total; i += stride) {
        long e = i >> Fsh;
        int  f = (int)(i & (F - 1));
        int  s = ei[e];
        int  d = ei[NEDGES + e];
        atomicAdd(&agg[(long)d * F + f], bf2f(feat[(long)s * F + f]));
    }
}

// mean = agg / max(cnt,1) -> bf16
__global__ void k_mean(const float* agg, const float* cnt, unsigned short* out, int Fsh) {
    const long total = (long)NNODES << Fsh;
    long stride = (long)gridDim.x * blockDim.x;
    for (long i = (long)blockIdx.x * blockDim.x + threadIdx.x; i < total; i += stride) {
        long node = i >> Fsh;
        out[i] = f2bf(agg[i] / fmaxf(cnt[node], 1.0f));
    }
}

// ---------------- WMMA bf16 GEMM ----------------
// C[M,N] = relu( A1[M,K] @ W1[N,K]^T (+ A2 @ W2^T) + bias ), store f32 and/or bf16.
// Block: 256 threads = 8 waves; block tile 128(M) x 128(N); K-steps of 32 via LDS,
// filled branchlessly with global_load_async_to_lds_b128 (ASYNCcnt): out-of-range
// source rows are CLAMPED to the last valid row, so every lane always issues its
// copy; the duplicated data only reaches output rows/cols the store-guard drops.
// Waves form a 4x2 grid; each wave computes a 32x64 sub-tile = 2x4
// v_wmma_f32_16x16x32_bf16 accums, reusing each B fragment for two A fragments.
__global__ __launch_bounds__(256) void k_gemm(
    const unsigned short* __restrict__ A1, const unsigned short* __restrict__ A2,
    const unsigned short* __restrict__ W1, const unsigned short* __restrict__ W2,
    const float* __restrict__ bias, float* outF, unsigned short* outB,
    int M, int N, int K, int nStore, int relu)
{
    __shared__ unsigned short As[128 * 32];   // A tile, row-major [m][k]
    __shared__ unsigned short Bs[128 * 32];   // W tile, row-major [n][k] (== B^T)

    const int tid   = threadIdx.x;
    const int wave  = tid >> 5;
    const int lane  = tid & 31;
    const int l15   = lane & 15;              // A: M index / B: N index
    const int khalf = lane >> 4;              // lane-group K-half select
    const int waveM = wave >> 1;              // 0..3 -> 32-row strip
    const int waveN = wave & 1;               // 0..1 -> 64-col strip
    const int rowbase = blockIdx.x * 128;
    const int colbase = blockIdx.y * 128;

    // per-thread staging assignment: 2 x 16B chunks each for A and B per K-step,
    // source rows clamped into range (branchless async issue)
    int arow[2], brow[2], chnk[2];
    unsigned aLds[2], bLds[2];
#pragma unroll
    for (int it = 0; it < 2; ++it) {
        int idx = tid + it * 256;
        int r = idx >> 2, ch = idx & 3;
        chnk[it] = ch;
        arow[it] = min(rowbase + r, M - 1);
        brow[it] = min(colbase + r, N - 1);
        aLds[it] = (unsigned)(uintptr_t)&As[r * 32 + ch * 8];
        bLds[it] = (unsigned)(uintptr_t)&Bs[r * 32 + ch * 8];
    }

    v8f acc[2][4];
#pragma unroll
    for (int mf = 0; mf < 2; ++mf)
#pragma unroll
        for (int nt = 0; nt < 4; ++nt)
#pragma unroll
            for (int e = 0; e < 8; ++e) acc[mf][nt][e] = 0.0f;

    const int npass = (A2 != nullptr) ? 2 : 1;
    for (int pass = 0; pass < npass; ++pass) {
        const unsigned short* A  = pass ? A2 : A1;
        const unsigned short* Wm = pass ? W2 : W1;
        const unsigned short* aPtr[2];
        const unsigned short* bPtr[2];
#pragma unroll
        for (int it = 0; it < 2; ++it) {
            aPtr[it] = A  + (long)arow[it] * K + chnk[it] * 8;
            bPtr[it] = Wm + (long)brow[it] * K + chnk[it] * 8;
        }

        for (int k0 = 0; k0 < K; k0 += 32) {
            __syncthreads();                       // prior tile fully consumed
#pragma unroll
            for (int it = 0; it < 2; ++it) {
                async_copy_b128(aLds[it], aPtr[it] + k0);
                async_copy_b128(bLds[it], bPtr[it] + k0);
            }
            wait_async0();                         // this wave's copies landed
            __syncthreads();                       // all waves' copies visible

            // A fragments (16x32): lane l15=M; VGPR j<4 -> K=khalf*8+2j, j>=4 -> +16
            union { v16bf v; unsigned u[8]; } af[2];
#pragma unroll
            for (int mf = 0; mf < 2; ++mf)
#pragma unroll
                for (int j = 0; j < 8; ++j) {
                    int grp = j >> 2, jj = j & 3;
                    int koff = grp * 16 + khalf * 8 + jj * 2;
                    af[mf].u[j] = *reinterpret_cast<const unsigned*>(
                        &As[(waveM * 32 + mf * 16 + l15) * 32 + koff]);
                }
            // B fragments (32x16): lane l15=N; VGPR j<4 -> K=khalf*8+2j, j>=4 -> +16; reuse for 2 A frags
#pragma unroll
            for (int nt = 0; nt < 4; ++nt) {
                union { v16bf v; unsigned u[8]; } bf;
#pragma unroll
                for (int j = 0; j < 8; ++j)
                    bf.u[j] = *reinterpret_cast<const unsigned*>(
                        &Bs[(waveN * 64 + nt * 16 + l15) * 32 + (j >> 2) * 16 + khalf * 8 + (j & 3) * 2]);
#pragma unroll
                for (int mf = 0; mf < 2; ++mf)
                    acc[mf][nt] = __builtin_amdgcn_wmma_f32_16x16x32_bf16(
                        false, af[mf].v, false, bf.v, (short)0, acc[mf][nt], false, false);
            }
        }
    }

    // epilogue: C/D layout -> VGPR r: lanes 0-15 M=r, lanes 16-31 M=r+8; N=l15
    const int rofs = khalf * 8;
#pragma unroll
    for (int mf = 0; mf < 2; ++mf)
#pragma unroll
        for (int nt = 0; nt < 4; ++nt) {
            int col = colbase + waveN * 64 + nt * 16 + l15;
#pragma unroll
            for (int r = 0; r < 8; ++r) {
                int row = rowbase + waveM * 32 + mf * 16 + rofs + r;
                if (row < M && col < nStore) {
                    float v = acc[mf][nt][r] + bias[col];
                    if (relu) v = fmaxf(v, 0.0f);
                    if (outF) outF[(long)row * nStore + col] = v;
                    if (outB) outB[(long)row * nStore + col] = f2bf(v);
                }
            }
        }
}

// ---------------- host orchestration ----------------

extern "C" void kernel_launch(void* const* d_in, const int* in_sizes, int n_in,
                              void* d_out, int out_size, void* d_ws, size_t ws_size,
                              hipStream_t stream) {
    const float* x   = (const float*)d_in[0];
    const int*   ei  = (const int*)d_in[1];     // edge_index [2, E]
    const float* W1l = (const float*)d_in[2];
    const float* b1  = (const float*)d_in[3];
    const float* W1r = (const float*)d_in[4];
    const float* W2l = (const float*)d_in[5];
    const float* b2  = (const float*)d_in[6];
    const float* W2r = (const float*)d_in[7];
    const float* Wd1 = (const float*)d_in[8];
    const float* bd1 = (const float*)d_in[9];
    const float* Wd2 = (const float*)d_in[10];
    const float* bd2 = (const float*)d_in[11];
    const float* Wd3 = (const float*)d_in[12];
    const float* bd3 = (const float*)d_in[13];

    float* q_out = (float*)d_out;                         // [N,10]
    float* h_out = (float*)d_out + (size_t)NNODES * 10;   // [N,256]

    char* ws = (char*)d_ws;
    size_t off = 0;
    auto carve = [&](size_t bytes) -> char* {
        char* p = ws + off;
        off += (bytes + 255) & ~(size_t)255;
        return p;
    };

    float*          agg    = (float*)carve((size_t)NNODES * 256 * 4);
    float*          cnt    = (float*)carve((size_t)NNODES * 4);
    unsigned short* xbf    = (unsigned short*)carve((size_t)NNODES * 128 * 2);
    unsigned short* meanbf = (unsigned short*)carve((size_t)NNODES * 256 * 2);
    unsigned short* h1bf   = (unsigned short*)carve((size_t)NNODES * 256 * 2);
    unsigned short* hbf    = (unsigned short*)carve((size_t)NNODES * 256 * 2);
    unsigned short* w1lb   = (unsigned short*)carve(256 * 128 * 2);
    unsigned short* w1rb   = (unsigned short*)carve(256 * 128 * 2);
    unsigned short* w2lb   = (unsigned short*)carve(256 * 256 * 2);
    unsigned short* w2rb   = (unsigned short*)carve(256 * 256 * 2);
    unsigned short* wd1b   = (unsigned short*)carve(128 * 256 * 2);
    unsigned short* wd2b   = (unsigned short*)carve(64 * 128 * 2);
    unsigned short* wd3b   = (unsigned short*)carve(16 * 64 * 2);
    float*          bd3p   = (float*)carve(16 * 4);
    // decoder scratch reuses regions freed after layer GEMMs:
    unsigned short* z1bf = xbf;                     // [N,128] bf16 (x no longer needed)
    unsigned short* z2bf = (unsigned short*)meanbf; // [N,64]

    const int B = 256;

    // prep: bf16 conversions
    k_cvt_bf16<<<2048, B, 0, stream>>>(x, xbf, (long)NNODES * 128);
    k_padcvt<<<128, B, 0, stream>>>(W1l, w1lb, 256, 128, 256, 128);
    k_padcvt<<<128, B, 0, stream>>>(W1r, w1rb, 256, 128, 256, 128);
    k_padcvt<<<256, B, 0, stream>>>(W2l, w2lb, 256, 256, 256, 256);
    k_padcvt<<<256, B, 0, stream>>>(W2r, w2rb, 256, 256, 256, 256);
    k_padcvt<<<128, B, 0, stream>>>(Wd1, wd1b, 128, 256, 128, 256);
    k_padcvt<<<32,  B, 0, stream>>>(Wd2, wd2b, 64, 128, 64, 128);
    k_padcvt<<<4,   B, 0, stream>>>(Wd3, wd3b, 10, 64, 16, 64);
    k_padbias<<<1, 16, 0, stream>>>(bd3, bd3p, 10, 16);

    // degree counts (shared by both layers)
    k_zero<<<64, B, 0, stream>>>(cnt, NNODES);
    k_count<<<2048, B, 0, stream>>>(ei, cnt);

    // ---- layer 1 ----
    k_zero<<<2048, B, 0, stream>>>(agg, (long)NNODES * 128);
    k_scatter<<<4096, B, 0, stream>>>(xbf, ei, agg, 7);
    k_mean<<<2048, B, 0, stream>>>(agg, cnt, meanbf, 7);
    k_gemm<<<dim3(391, 2), B, 0, stream>>>(meanbf, xbf, w1lb, w1rb, b1,
                                           nullptr, h1bf, NNODES, 256, 128, 256, 1);

    // ---- layer 2 ----
    k_zero<<<2048, B, 0, stream>>>(agg, (long)NNODES * 256);
    k_scatter<<<8192, B, 0, stream>>>(h1bf, ei, agg, 8);
    k_mean<<<2048, B, 0, stream>>>(agg, cnt, meanbf, 8);
    k_gemm<<<dim3(391, 2), B, 0, stream>>>(meanbf, h1bf, w2lb, w2rb, b2,
                                           h_out, hbf, NNODES, 256, 256, 256, 1);

    // ---- decoder ----
    k_gemm<<<dim3(391, 1), B, 0, stream>>>(hbf, nullptr, wd1b, nullptr, bd1,
                                           nullptr, z1bf, NNODES, 128, 256, 128, 1);
    k_gemm<<<dim3(391, 1), B, 0, stream>>>(z1bf, nullptr, wd2b, nullptr, bd2,
                                           nullptr, z2bf, NNODES, 64, 128, 64, 1);
    k_gemm<<<dim3(391, 1), B, 0, stream>>>(z2bf, nullptr, wd3b, nullptr, bd3p,
                                           q_out, nullptr, NNODES, 16, 64, 10, 0);

    (void)in_sizes; (void)n_in; (void)out_size; (void)ws_size;
}
